// HimNet_multimode_v3_7645041787314
// MI455X (gfx1250) — matrix operand order repaired and checked by reading, not compiled, not run on hardware
//
#include <hip/hip_runtime.h>
#include <math.h>

typedef _Float16 h8  __attribute__((ext_vector_type(8)));
typedef _Float16 h16 __attribute__((ext_vector_type(16)));
typedef float    f8  __attribute__((ext_vector_type(8)));

// ---------------- problem dims ----------------
constexpr int kB    = 8;
constexpr int kN    = 1000;   // nodes
constexpr int kNP   = 1024;   // padded nodes
constexpr int kCIN  = 16;
constexpr int kH    = 64;
constexpr int kXIN  = 80;     // CIN + H
constexpr int kKCP  = 256;    // padded K*XIN (240 -> 256)
constexpr int kD    = 16;
constexpr int kOG   = 128;    // gate out dim (2H)
constexpr int kOU   = 64;     // update out dim (H)
constexpr int kCOLS = 640;    // B * XIN
constexpr int kROWS = 8192;   // NP * B

// ---------------- workspace layout (bytes) ----------------
constexpr size_t OFF_WG16T = 0;                                   // [16][128][256] f16
constexpr size_t SZ_WG16T  = (size_t)16 * 128 * 256 * 2;
constexpr size_t OFF_WU16T = OFF_WG16T + SZ_WG16T;                // [16][64][256] f16
constexpr size_t SZ_WU16T  = (size_t)16 * 64 * 256 * 2;
constexpr size_t OFF_MODES = OFF_WU16T + SZ_WU16T;
// per-mode region
constexpr size_t MO_S16   = 0;                                    // [1024][1024] f16
constexpr size_t MO_XST   = MO_S16  + (size_t)kNP * kNP * 2;      // [640][1024] f16 (col-major B op)
constexpr size_t MO_T1T   = MO_XST  + (size_t)kCOLS * kNP * 2;    // [640][1024] f16
constexpr size_t MO_XGG   = MO_T1T  + (size_t)kCOLS * kNP * 2;    // [8192][256] f16 (gate x_g)
constexpr size_t MO_XGU   = MO_XGG  + (size_t)kROWS * kKCP * 2;   // [8192][256] f16 (update x_g)
constexpr size_t MO_ZR    = MO_XGU  + (size_t)kROWS * kKCP * 2;   // [8192][128] f16 (z|r)
constexpr size_t MO_E32   = MO_ZR   + (size_t)kROWS * 128 * 2;    // [1024][16] f32 emb*me
constexpr size_t MODE_SZ  = MO_E32  + (size_t)kNP * 16 * 4;

// ---------------- fragment helpers (wave32 WMMA layouts) ----------------
__device__ __forceinline__ h16 cat16(h8 a, h8 b) {
  return __builtin_shufflevector(a, b, 0,1,2,3,4,5,6,7,8,9,10,11,12,13,14,15);
}
// A operand 16x32 f16: row-major source, row = per-lane ptr; per-lane halves:
// k = k0 + (lane/16)*16 + h
__device__ __forceinline__ h16 ldA(const _Float16* row, int k0, int lane) {
  int ko = k0 + ((lane >> 4) << 4);
  h8 a0 = *(const h8*)(row + ko);
  h8 a1 = *(const h8*)(row + ko + 8);
  return cat16(a0, a1);
}
// B operand 32x16 f16: column-major source (col ptr = &B[n][0], k contiguous);
// per-lane halves: k = k0 + (lane/16)*16 + h, n = lane%16
__device__ __forceinline__ h16 ldB(const _Float16* col, int k0, int lane) {
  int ko = k0 + ((lane >> 4) << 4);
  h8 b0 = *(const h8*)(col + ko);
  h8 b1 = *(const h8*)(col + ko + 8);
  return cat16(b0, b1);
}
__device__ __forceinline__ f8 wmma16(h16 a, h16 b, f8 c) {
  return __builtin_amdgcn_wmma_f32_16x16x32_f16(false, a, false, b, (short)0, c, false, false);
}

// ---------------- prep kernels ----------------
// Wg [16][240][128] f32 -> Wg16T [d][o][256] f16 (i fastest, padded);  same for Wu
__global__ void prep_pools(const float* __restrict__ Wg, const float* __restrict__ Wu,
                           char* __restrict__ ws) {
  size_t idx = (size_t)blockIdx.x * 256 + threadIdx.x;
  _Float16* G = (_Float16*)(ws + OFF_WG16T);
  _Float16* U = (_Float16*)(ws + OFF_WU16T);
  constexpr size_t NG = (size_t)16 * 128 * 256;
  constexpr size_t NU = (size_t)16 * 64 * 256;
  if (idx < NG) {
    int d = (int)(idx / (128 * 256));
    int o = (int)((idx / 256) % 128);
    int i = (int)(idx & 255);
    float v = (i < 240) ? Wg[((size_t)d * 240 + i) * 128 + o] : 0.0f;
    G[idx] = (_Float16)v;
  } else if (idx < NG + NU) {
    size_t k = idx - NG;
    int d = (int)(k / (64 * 256));
    int o = (int)((k / 256) % 64);
    int i = (int)(k & 255);
    float v = (i < 240) ? Wu[((size_t)d * 240 + i) * 64 + o] : 0.0f;
    U[k] = (_Float16)v;
  }
}

// embm[n][d] = emb[n][d] * mode_embedding[m][d]  (f32)
__global__ void prep_embm(const float* e1, const float* e2, const float* e3,
                          const float* __restrict__ me, char* __restrict__ ws) {
  int m = blockIdx.z;
  char* wsm = ws + OFF_MODES + (size_t)m * MODE_SZ;
  float* E32 = (float*)(wsm + MO_E32);
  int idx = blockIdx.x * 256 + threadIdx.x;        // 1024*16
  int n = idx >> 4, d = idx & 15;
  const float* emb = (m == 0) ? e1 : (m == 1) ? e2 : e3;
  float v = (n < kN) ? emb[n * 16 + d] * me[m * 16 + d] : 0.0f;
  E32[idx] = v;
}

// support f32 [1000][1000] -> f16 [1024][1024] zero-padded
__global__ void prep_s16(const float* s1, const float* s2, const float* s3,
                         char* __restrict__ ws) {
  int m = blockIdx.z;
  _Float16* S = (_Float16*)(ws + OFF_MODES + (size_t)m * MODE_SZ + MO_S16);
  size_t idx = (size_t)blockIdx.x * 256 + threadIdx.x;   // 1024*1024
  int r = (int)(idx >> 10), c = (int)(idx & 1023);
  const float* sup = (m == 0) ? s1 : (m == 1) ? s2 : s3;
  float v = (r < kN && c < kN) ? sup[(size_t)r * kN + c] : 0.0f;
  S[idx] = (_Float16)v;
}

// build gate x_g term0 = concat(x,state) into XG_G[row=(n*8+b)][i] and XsT[j=(b*80+c)][n]
__global__ void prep_xg_gate(const float* x1, const float* x2, const float* x3,
                             const float* t1, const float* t2, const float* t3,
                             char* __restrict__ ws) {
  int m = blockIdx.z;
  char* wsm = ws + OFF_MODES + (size_t)m * MODE_SZ;
  _Float16* XG  = (_Float16*)(wsm + MO_XGG);
  _Float16* XsT = (_Float16*)(wsm + MO_XST);
  size_t idx = (size_t)blockIdx.x * 256 + threadIdx.x;   // 8192*256
  int row = (int)(idx >> 8), i = (int)(idx & 255);
  int n = row >> 3, b = row & 7;
  const float* x  = (m == 0) ? x1 : (m == 1) ? x2 : x3;
  const float* st = (m == 0) ? t1 : (m == 1) ? t2 : t3;
  float v = 0.0f;
  if (n < kN) {
    if (i < kCIN)      v = x[((size_t)b * kN + n) * kCIN + i];
    else if (i < kXIN) v = st[((size_t)b * kN + n) * kH + (i - kCIN)];
  }
  XG[idx] = (_Float16)v;
  if (i < kXIN) XsT[((size_t)(b * kXIN + i)) * kNP + n] = (_Float16)v;
}

// build update x_g term0 = concat(x, z*state); z = ZR[:, :64]
__global__ void prep_xg_upd(const float* x1, const float* x2, const float* x3,
                            const float* t1, const float* t2, const float* t3,
                            char* __restrict__ ws) {
  int m = blockIdx.z;
  char* wsm = ws + OFF_MODES + (size_t)m * MODE_SZ;
  _Float16* XG  = (_Float16*)(wsm + MO_XGU);
  _Float16* XsT = (_Float16*)(wsm + MO_XST);          // reuse buffer for xs2^T
  const _Float16* ZR = (const _Float16*)(wsm + MO_ZR);
  size_t idx = (size_t)blockIdx.x * 256 + threadIdx.x;
  int row = (int)(idx >> 8), i = (int)(idx & 255);
  int n = row >> 3, b = row & 7;
  const float* x  = (m == 0) ? x1 : (m == 1) ? x2 : x3;
  const float* st = (m == 0) ? t1 : (m == 1) ? t2 : t3;
  float v = 0.0f;
  if (n < kN) {
    if (i < kCIN) {
      v = x[((size_t)b * kN + n) * kCIN + i];
    } else if (i < kXIN) {
      float z = (float)ZR[(size_t)row * 128 + (i - kCIN)];
      v = z * st[((size_t)b * kN + n) * kH + (i - kCIN)];
    }
  }
  XG[idx] = (_Float16)v;
  if (i < kXIN) XsT[((size_t)(b * kXIN + i)) * kNP + n] = (_Float16)v;
}

// ---------------- spmm: C[n, j] = S[n, m] * B[m, j]  (M=1024, K=1024, 640 cols) ----------------
// VARIANT 0: B = XsT, write T1 (f16) to T1T[j][n] and XG[row][80+c]
// VARIANT 1: B = T1T, val = 2*acc - Xs, write XG[row][160+c]
template <int VARIANT>
__global__ __launch_bounds__(128) void spmm_kernel(char* __restrict__ ws,
                                                   unsigned long long xgOff) {
  int m = blockIdx.z;
  char* wsm = ws + OFF_MODES + (size_t)m * MODE_SZ;
  const _Float16* S   = (const _Float16*)(wsm + MO_S16);
  const _Float16* Bt  = (const _Float16*)(wsm + (VARIANT ? MO_T1T : MO_XST));
  const _Float16* Sub = (const _Float16*)(wsm + MO_XST);
  _Float16* OutT = (_Float16*)(wsm + MO_T1T);
  _Float16* XG   = (_Float16*)(wsm + (size_t)xgOff);
  int lane = threadIdx.x & 31;
  int wv   = threadIdx.x >> 5;
  int rt = blockIdx.x * 4 + wv;     // 0..63  (node tiles)
  int cg = blockIdx.y;              // 0..9   (groups of 4 col-tiles)

  const _Float16* arow = S + (size_t)(rt * 16 + (lane & 15)) * kNP;
  const _Float16* bcol[4];
#pragma unroll
  for (int t = 0; t < 4; ++t)
    bcol[t] = Bt + (size_t)((cg * 4 + t) * 16 + (lane & 15)) * kNP;

  f8 acc[4] = {};
  // software pipeline: fragments for k-tile kt are loaded during k-tile kt-1's wmmas
  h16 a = ldA(arow, 0, lane);
  h16 b[4];
#pragma unroll
  for (int t = 0; t < 4; ++t) b[t] = ldB(bcol[t], 0, lane);

  for (int kt = 0; kt < 32; ++kt) {
    h16 an = a;
    h16 bn[4];
#pragma unroll
    for (int t = 0; t < 4; ++t) bn[t] = b[t];
    if (kt < 31) {
      int k0n = (kt + 1) * 32;
      an = ldA(arow, k0n, lane);
#pragma unroll
      for (int t = 0; t < 4; ++t) bn[t] = ldB(bcol[t], k0n, lane);
    }
#pragma unroll
    for (int t = 0; t < 4; ++t) acc[t] = wmma16(a, b[t], acc[t]);
    a = an;
#pragma unroll
    for (int t = 0; t < 4; ++t) b[t] = bn[t];
  }

  constexpr int termoff = VARIANT ? 160 : 80;
#pragma unroll
  for (int t = 0; t < 4; ++t) {
    int j_g = (cg * 4 + t) * 16 + (lane & 15);
    int bb = j_g / kXIN;
    int cc = j_g - bb * kXIN;
#pragma unroll
    for (int rr = 0; rr < 8; ++rr) {
      int n_g = rt * 16 + ((lane >> 4) << 3) + rr;
      float v = acc[t][rr];
      if (VARIANT) v = 2.0f * v - (float)Sub[(size_t)j_g * kNP + n_g];
      _Float16 hv = (_Float16)v;
      if (!VARIANT) OutT[(size_t)j_g * kNP + n_g] = hv;
      XG[((size_t)n_g * 8 + bb) * kKCP + termoff + cc] = hv;
    }
  }
}

// ---------------- hypernet apply: out[row,o] = sum_d embm[n,d] * (Xg @ Wp[d])[row,o] ----------
// A fragments register-resident and untouched (no WMMA->VALU WAR hazard on A);
// per-d partial accumulated with a pure WMMA chain, then folded into the f32
// accumulator with the per-(node,d) scale: acc += s * inner  (scale in f32).
__device__ __forceinline__ f8 hyper_mm(const _Float16* __restrict__ XG,
                                       const _Float16* __restrict__ Wt,
                                       const float* __restrict__ E32,
                                       int Odim, int rt, int ot, int lane) {
  const _Float16* ap = XG + (size_t)(rt * 16 + (lane & 15)) * kKCP;
  h16 A[8];
#pragma unroll
  for (int kt = 0; kt < 8; ++kt) A[kt] = ldA(ap, kt * 32, lane);
  int nE = rt * 2 + (lane >> 4);     // node of this lane's 8 D-rows
  int oL = ot * 16 + (lane & 15);

  f8 acc = {};
  h16 b = ldB(Wt + (size_t)oL * kKCP, 0, lane);   // (d=0, kt=0)
  for (int d = 0; d < kD; ++d) {
    f8 inner = {};
#pragma unroll
    for (int kt = 0; kt < 8; ++kt) {
      // prefetch next (d,kt) fragment
      int ktn = kt + 1, dn = d;
      if (ktn == 8) { ktn = 0; dn = d + 1; }
      h16 bnx = b;
      if (dn < kD)
        bnx = ldB(Wt + ((size_t)dn * Odim + oL) * kKCP, ktn * 32, lane);
      inner = wmma16(A[kt], b, inner);
      b = bnx;
    }
    float s = E32[nE * 16 + d];
#pragma unroll
    for (int e = 0; e < 8; ++e) acc[e] += s * inner[e];
  }
  return acc;
}

__global__ __launch_bounds__(128) void apply_gate(char* __restrict__ ws,
                                                  const float* __restrict__ Bg) {
  int m = blockIdx.z;
  char* wsm = ws + OFF_MODES + (size_t)m * MODE_SZ;
  const _Float16* XG  = (const _Float16*)(wsm + MO_XGG);
  const _Float16* Wt  = (const _Float16*)(ws + OFF_WG16T);
  const float*    E32 = (const float*)(wsm + MO_E32);
  _Float16* ZR = (_Float16*)(wsm + MO_ZR);
  int lane = threadIdx.x & 31;
  int wv   = threadIdx.x >> 5;
  int rt = blockIdx.x * 4 + wv;     // 0..511 (row tiles of (n,b))
  int ot = blockIdx.y;              // 0..7

  f8 acc = hyper_mm(XG, Wt, E32, kOG, rt, ot, lane);

  int oL   = ot * 16 + (lane & 15);
  int nE   = rt * 2 + (lane >> 4);
  int rowb = rt * 16 + ((lane >> 4) << 3);
  float bias = 0.0f;
#pragma unroll
  for (int d = 0; d < kD; ++d) bias += E32[nE * 16 + d] * Bg[d * kOG + oL];
#pragma unroll
  for (int rr = 0; rr < 8; ++rr) {
    float t = acc[rr] + bias;
    float v = 1.0f / (1.0f + __expf(-t));
    ZR[(size_t)(rowb + rr) * 128 + oL] = (_Float16)v;
  }
}

__global__ __launch_bounds__(128) void apply_upd(char* __restrict__ ws,
                                                 const float* __restrict__ Bu,
                                                 const float* t1, const float* t2, const float* t3,
                                                 float* __restrict__ out) {
  int m = blockIdx.z;
  char* wsm = ws + OFF_MODES + (size_t)m * MODE_SZ;
  const _Float16* XG  = (const _Float16*)(wsm + MO_XGU);
  const _Float16* Wt  = (const _Float16*)(ws + OFF_WU16T);
  const float*    E32 = (const float*)(wsm + MO_E32);
  const _Float16* ZR  = (const _Float16*)(wsm + MO_ZR);
  const float* st = (m == 0) ? t1 : (m == 1) ? t2 : t3;
  int lane = threadIdx.x & 31;
  int wv   = threadIdx.x >> 5;
  int rt = blockIdx.x * 4 + wv;
  int ot = blockIdx.y;              // 0..3

  f8 acc = hyper_mm(XG, Wt, E32, kOU, rt, ot, lane);

  int oL   = ot * 16 + (lane & 15);
  int nE   = rt * 2 + (lane >> 4);
  int rowb = rt * 16 + ((lane >> 4) << 3);
  float bias = 0.0f;
#pragma unroll
  for (int d = 0; d < kD; ++d) bias += E32[nE * 16 + d] * Bu[d * kOU + oL];
#pragma unroll
  for (int rr = 0; rr < 8; ++rr) {
    int row = rowb + rr;                 // row = n*8 + b, b == rr
    float t  = acc[rr] + bias;
    float hc = tanhf(t);
    float rg = (float)ZR[(size_t)row * 128 + 64 + oL];
    if (nE < kN) {
      float sv = st[((size_t)rr * kN + nE) * kH + oL];
      out[(size_t)m * kB * kN * kH + ((size_t)rr * kN + nE) * kH + oL] =
          rg * sv + (1.0f - rg) * hc;
    }
  }
}

// ---------------- launch ----------------
extern "C" void kernel_launch(void* const* d_in, const int* in_sizes, int n_in,
                              void* d_out, int out_size, void* d_ws, size_t ws_size,
                              hipStream_t stream) {
  const float* x1 = (const float*)d_in[0];
  const float* x2 = (const float*)d_in[1];
  const float* x3 = (const float*)d_in[2];
  const float* s1 = (const float*)d_in[3];
  const float* s2 = (const float*)d_in[4];
  const float* s3 = (const float*)d_in[5];
  const float* g1 = (const float*)d_in[6];
  const float* g2 = (const float*)d_in[7];
  const float* g3 = (const float*)d_in[8];
  const float* e1 = (const float*)d_in[9];
  const float* e2 = (const float*)d_in[10];
  const float* e3 = (const float*)d_in[11];
  const float* me = (const float*)d_in[12];
  const float* Wg = (const float*)d_in[13];
  const float* Bg = (const float*)d_in[14];
  const float* Wu = (const float*)d_in[15];
  const float* Bu = (const float*)d_in[16];
  (void)in_sizes; (void)n_in; (void)out_size; (void)ws_size;
  char*  ws  = (char*)d_ws;
  float* out = (float*)d_out;

  dim3 b256(256), b128(128);
  prep_pools<<<dim3(3072), b256, 0, stream>>>(Wg, Wu, ws);
  prep_embm<<<dim3(64, 1, 3), b256, 0, stream>>>(e1, e2, e3, me, ws);
  prep_s16<<<dim3(4096, 1, 3), b256, 0, stream>>>(g1, g2, g3, ws);
  prep_xg_gate<<<dim3(8192, 1, 3), b256, 0, stream>>>(x1, x2, x3, s1, s2, s3, ws);
  // gate GCN: T1 then T2 (Chebyshev), then hypernet apply + sigmoid
  spmm_kernel<0><<<dim3(16, 10, 3), b128, 0, stream>>>(ws, (unsigned long long)MO_XGG);
  spmm_kernel<1><<<dim3(16, 10, 3), b128, 0, stream>>>(ws, (unsigned long long)MO_XGG);
  apply_gate<<<dim3(128, 8, 3), b128, 0, stream>>>(ws, Bg);
  // update GCN on concat(x, z*state)
  prep_xg_upd<<<dim3(8192, 1, 3), b256, 0, stream>>>(x1, x2, x3, s1, s2, s3, ws);
  spmm_kernel<0><<<dim3(16, 10, 3), b128, 0, stream>>>(ws, (unsigned long long)MO_XGU);
  spmm_kernel<1><<<dim3(16, 10, 3), b128, 0, stream>>>(ws, (unsigned long long)MO_XGU);
  apply_upd<<<dim3(128, 4, 3), b128, 0, stream>>>(ws, Bu, s1, s2, s3, out);
}
